// RNNClassifier_74062416052963
// MI455X (gfx1250) — matrix operand order repaired and checked by reading, not compiled, not run on hardware
//
#include <hip/hip_runtime.h>
#include <math.h>

// ---- CDNA5 WMMA types -------------------------------------------------------
typedef _Float16 v16h __attribute__((ext_vector_type(16)));
typedef _Float16 v8h  __attribute__((ext_vector_type(8)));
typedef float    v8f  __attribute__((ext_vector_type(8)));

// Problem constants (match reference setup_inputs()).
constexpr int B  = 512;
constexpr int T  = 1024;
constexpr int I  = 32;
constexpr int H1 = 128;
constexpr int H2 = 64;
constexpr int NC = 10;
constexpr int BT = 16;     // batch rows per workgroup (one WMMA M tile)

// D = A(16x32 f16) * B(32x16 f16) + C(16x16 f32)
__device__ __forceinline__ v8f wmma16(v16h a, v16h b, v8f c) {
  return __builtin_amdgcn_wmma_f32_16x16x32_f16(
      /*neg_a=*/false, a, /*neg_b=*/false, b,
      /*c_mod=*/(short)0, c, /*reuse_a=*/false, /*reuse_b=*/false);
}

// Batched hardware tanh (CDNA5 TRANS32 op).  Each v_tanh_f32 is the
// independent-op hazard filler for the previous one; only the last needs a
// v_nop.  Early-clobber outputs guarantee no dest aliases a later input.
__device__ __forceinline__ v8f tanh8(v8f x) {
  float o0, o1, o2, o3, o4, o5, o6, o7;
  asm("v_tanh_f32 %0, %8\n\t"
      "v_tanh_f32 %1, %9\n\t"
      "v_tanh_f32 %2, %10\n\t"
      "v_tanh_f32 %3, %11\n\t"
      "v_tanh_f32 %4, %12\n\t"
      "v_tanh_f32 %5, %13\n\t"
      "v_tanh_f32 %6, %14\n\t"
      "v_tanh_f32 %7, %15\n\t"
      "v_nop"
      : "=&v"(o0), "=&v"(o1), "=&v"(o2), "=&v"(o3),
        "=&v"(o4), "=&v"(o5), "=&v"(o6), "=&v"(o7)
      : "v"(x[0]), "v"(x[1]), "v"(x[2]), "v"(x[3]),
        "v"(x[4]), "v"(x[5]), "v"(x[6]), "v"(x[7]));
  v8f r;
  r[0] = o0; r[1] = o1; r[2] = o2; r[3] = o3;
  r[4] = o4; r[5] = o5; r[6] = o6; r[7] = o7;
  return r;
}

// A fragment, from a per-lane pointer p = &buf[m*kstride + hi*8] derived
// DIRECTLY from the __shared__ array (keeps addrspace(3) -> ds_load_b128).
// CDNA5 16-bit A layout: half h -> k = kb + hi*8 + (h<8 ? h : 8+h); halves
// 0..7 and 8..15 are each 16B-contiguous -> two ds_load_b128 with immediate
// offsets off one address register.
__device__ __forceinline__ v16h frag_a_p(const _Float16* p, int kb) {
  v8h lo = *(const v8h*)(p + kb);
  v8h up = *(const v8h*)(p + kb + 16);
  return __builtin_shufflevector(lo, up, 0, 1, 2, 3, 4, 5, 6, 7,
                                         8, 9, 10, 11, 12, 13, 14, 15);
}

// B fragment (Kx16 tile of W^T, W stored row-major [N][kstride] f16 in LDS).
// CDNA5 16-bit B layout: lane n=lane&15, hi=lane>>4; half h -> k = kb + hi*8 + (h<8 ? h : 8+h).
__device__ __forceinline__ v16h frag_b(const _Float16* base, int kstride,
                                       int n0, int kb, int lane) {
  const int n  = lane & 15;
  const int hi = (lane >> 4) & 1;
  const _Float16* p = base + (n0 + n) * kstride + kb + hi * 8;
  v8h lo = *(const v8h*)(p);
  v8h up = *(const v8h*)(p + 16);
  return __builtin_shufflevector(lo, up, 0, 1, 2, 3, 4, 5, 6, 7,
                                         8, 9, 10, 11, 12, 13, 14, 15);
}

__global__ __launch_bounds__(256)
void rnn2_fused_kernel(const float* __restrict__ x,
                       const float* __restrict__ Wih1,
                       const float* __restrict__ Whh1,
                       const float* __restrict__ bih1,
                       const float* __restrict__ bhh1,
                       const float* __restrict__ Wih2,
                       const float* __restrict__ Whh2,
                       const float* __restrict__ bih2,
                       const float* __restrict__ bhh2,
                       const float* __restrict__ Wfc,
                       const float* __restrict__ bfc,
                       float* __restrict__ out) {
  // Weights live in LDS as f16, row-major [n][k] (== B^T layout for h@W^T).
  __shared__ _Float16 s_wih1[H1 * I];    //  8 KB
  __shared__ _Float16 s_whh1[H1 * H1];   // 32 KB
  __shared__ _Float16 s_wih2[H2 * H1];   // 16 KB
  __shared__ _Float16 s_whh2[H2 * H2];   //  8 KB
  __shared__ float    s_b1[H1];
  __shared__ float    s_b2[H2];
  __shared__ _Float16 s_x[BT * I];       //  1 KB  (x_t tile, f16)
  __shared__ _Float16 s_h1[2][BT * H1];  //  8 KB  (ping-pong)
  __shared__ _Float16 s_h2[2][BT * H2];  //  4 KB  (ping-pong)

  const int tid  = threadIdx.x;
  const int lane = tid & 31;
  const int wave = tid >> 5;         // 0..7
  const int b0   = blockIdx.x * BT;  // batch-tile origin

  // ---- Phase 0: stage weights/biases into LDS (f16), zero h state ----------
  for (int i = tid; i < H1 * I;  i += 256) s_wih1[i] = (_Float16)Wih1[i];
  for (int i = tid; i < H1 * H1; i += 256) s_whh1[i] = (_Float16)Whh1[i];
  for (int i = tid; i < H2 * H1; i += 256) s_wih2[i] = (_Float16)Wih2[i];
  for (int i = tid; i < H2 * H2; i += 256) s_whh2[i] = (_Float16)Whh2[i];
  for (int i = tid; i < H1; i += 256) s_b1[i] = bih1[i] + bhh1[i];
  for (int i = tid; i < H2; i += 256) s_b2[i] = bih2[i] + bhh2[i];
  for (int i = tid; i < BT * H1; i += 256) s_h1[0][i] = (_Float16)0.0f;
  for (int i = tid; i < BT * H2; i += 256) s_h2[0][i] = (_Float16)0.0f;
  __syncthreads();

  // ---- Phase 1: persistent B fragments in VGPRs for the whole scan ---------
  const int n1 = wave * 16;  // this wave's layer-1 N tile
  v16h bF_ih1 = frag_b(s_wih1, I, n1, 0, lane);
  v16h bF_hh1[4];
#pragma unroll
  for (int kc = 0; kc < 4; ++kc)
    bF_hh1[kc] = frag_b(s_whh1, H1, n1, kc * 32, lane);

  v8f cb1;
  {
    const float bv = s_b1[n1 + (lane & 15)];
#pragma unroll
    for (int r = 0; r < 8; ++r) cb1[r] = bv;
  }

  v16h bF_ih2[4];
  v16h bF_hh2[2];
  v8f  cb2;
  if (wave < 4) {                    // waves 0..3 own layer-2 N tiles
    const int n2 = wave * 16;
#pragma unroll
    for (int kc = 0; kc < 4; ++kc)
      bF_ih2[kc] = frag_b(s_wih2, H1, n2, kc * 32, lane);
#pragma unroll
    for (int kc = 0; kc < 2; ++kc)
      bF_hh2[kc] = frag_b(s_whh2, H2, n2, kc * 32, lane);
    const float bv = s_b2[n2 + (lane & 15)];
#pragma unroll
    for (int r = 0; r < 8; ++r) cb2[r] = bv;
  }

  // Per-lane loop-invariant index bases (kept as integers so every pointer is
  // derived directly from the __shared__ arrays -> stays addrspace(3)/DS).
  const int nlo  = lane & 15;
  const int hi8  = ((lane >> 4) & 1) * 8;
  const int aB1  = nlo * H1 + hi8;          // A-frag read base in s_h1[*]
  const int aB2  = nlo * H2 + hi8;          // A-frag read base in s_h2[*]
  const int sB1  = hi8 * H1 + n1 + nlo;     // C store base in s_h1[*] (m=r+hi8)
  const int sB2  = hi8 * H2 + wave * 16 + nlo;

  // x-staging role of this thread: row 0..15, two consecutive f32 columns.
  const int row  = tid >> 4;
  const int col2 = (tid & 15) * 2;
  const long xbase = (long)(b0 + row) * T * I + col2;
  const _Float16* axrd = &s_x[nlo * I + hi8];

  // ---- Phase 2: fused two-layer recurrence over T steps --------------------
  for (int t = 0; t < T; ++t) {
    // Stage x[:, t, :] tile (f32 -> f16) into LDS; prefetch 16 steps ahead.
    {
      const float2 v = *(const float2*)(x + xbase + (long)t * I);
      s_x[row * I + col2]     = (_Float16)v.x;
      s_x[row * I + col2 + 1] = (_Float16)v.y;
      if (t + 16 < T && (tid & 15) == 0)
        __builtin_prefetch(x + xbase + (long)(t + 16) * I, 0, 1);
    }
    __syncthreads();  // x_t visible; h1/h2 writes from step t-1 visible

    const int cur = t & 1;
    const int nxt = cur ^ 1;

    // Layer 1: h1' = tanh(x_t @ Wih1^T + h1 @ Whh1^T + b1), N-tile per wave.
    // Issue all A-fragment LDS loads first so ds latency overlaps the WMMA
    // chain (partial s_wait_dscnt instead of full stalls).
    v16h aX = frag_a_p(axrd, 0);
    v16h aH[4];
    {
      const _Float16* ra = &s_h1[cur][aB1];
#pragma unroll
      for (int kc = 0; kc < 4; ++kc)
        aH[kc] = frag_a_p(ra, kc * 32);
    }

    v8f acc = cb1;
    acc = wmma16(aX, bF_ih1, acc);
#pragma unroll
    for (int kc = 0; kc < 4; ++kc)
      acc = wmma16(aH[kc], bF_hh1[kc], acc);

    {
      const v8f th = tanh8(acc);
      _Float16* hw = &s_h1[nxt][sB1];
#pragma unroll
      for (int r = 0; r < 8; ++r)
        hw[r * H1] = (_Float16)th[r];   // ds_store_b16, immediate offsets
    }

    __syncthreads();  // h1' visible to layer 2

    // Layer 2: h2' = tanh(h1' @ Wih2^T + h2 @ Whh2^T + b2) on waves 0..3.
    if (wave < 4) {
      v16h a2[6];
      {
        const _Float16* ra1 = &s_h1[nxt][aB1];   // h1' lives in buffer nxt
        const _Float16* ra2 = &s_h2[cur][aB2];
#pragma unroll
        for (int kc = 0; kc < 4; ++kc)
          a2[kc] = frag_a_p(ra1, kc * 32);
#pragma unroll
        for (int kc = 0; kc < 2; ++kc)
          a2[4 + kc] = frag_a_p(ra2, kc * 32);
      }

      v8f acc2 = cb2;
#pragma unroll
      for (int kc = 0; kc < 4; ++kc)
        acc2 = wmma16(a2[kc], bF_ih2[kc], acc2);
#pragma unroll
      for (int kc = 0; kc < 2; ++kc)
        acc2 = wmma16(a2[4 + kc], bF_hh2[kc], acc2);

      const v8f th = tanh8(acc2);
      _Float16* hw = &s_h2[nxt][sB2];
#pragma unroll
      for (int r = 0; r < 8; ++r)
        hw[r * H2] = (_Float16)th[r];
    }
    // next iteration's first barrier publishes h2'
  }
  __syncthreads();

  // ---- Phase 3: FC head  logits[16,10] = h2_final @ Wfc^T + bfc ------------
  // After step t the new state is written to buffer (t+1)&1, so after
  // t = T-1 the final state lives in buffer T&1 == 0.
  if (tid < BT * NC) {
    const int m = tid / NC;
    const int c = tid % NC;
    float s = bfc[c];
#pragma unroll 8
    for (int k = 0; k < H2; ++k)
      s += (float)s_h2[0][m * H2 + k] * Wfc[c * H2 + k];
    out[(long)(b0 + m) * NC + c] = s;
  }
}

extern "C" void kernel_launch(void* const* d_in, const int* in_sizes, int n_in,
                              void* d_out, int out_size, void* d_ws, size_t ws_size,
                              hipStream_t stream) {
  (void)in_sizes; (void)n_in; (void)out_size; (void)d_ws; (void)ws_size;
  const float* x    = (const float*)d_in[0];
  const float* Wih1 = (const float*)d_in[1];
  const float* Whh1 = (const float*)d_in[2];
  const float* bih1 = (const float*)d_in[3];
  const float* bhh1 = (const float*)d_in[4];
  const float* Wih2 = (const float*)d_in[5];
  const float* Whh2 = (const float*)d_in[6];
  const float* bih2 = (const float*)d_in[7];
  const float* bhh2 = (const float*)d_in[8];
  const float* Wfc  = (const float*)d_in[9];
  const float* bfc  = (const float*)d_in[10];
  float* out = (float*)d_out;

  dim3 grid(B / BT);   // 32 workgroups: full tile-level parallelism of the scan
  dim3 block(256);     // 8 wave32 waves: 8 layer-1 N tiles, 4 layer-2 N tiles
  rnn2_fused_kernel<<<grid, block, 0, stream>>>(
      x, Wih1, Whh1, bih1, bhh1, Wih2, Whh2, bih2, bhh2, Wfc, bfc, out);
}
